// STCGNN_68556267979445
// MI455X (gfx1250) — compile-verified, matrix operand order checked by reading, not run on hardware
//
#include <hip/hip_runtime.h>
#include <math.h>

typedef __attribute__((ext_vector_type(16))) __bf16 v16bf;
typedef __attribute__((ext_vector_type(8)))  float  v8f;

static constexpr int B_ = 4, T_ = 6, N_ = 512, C_ = 8;
static constexpr int HID_ = 64, FEAT_ = 16;
static constexpr int ROWS_ = B_ * N_ * C_;              // 16384
static constexpr int L_ENC = 1 + HID_ + FEAT_;          // 81
static constexpr int L_DEC = HID_ + HID_ + FEAT_;       // 144
static constexpr int CLP_ENC = 704;                     // pad of 648 to /64
static constexpr int CLP_DEC = 1152;                    // already /64
static constexpr int KJL_ENC = 6 * L_ENC;               // 486
static constexpr int KJLP_ENC = 512;
static constexpr int KJL_DEC = 6 * L_DEC;               // 864
static constexpr int KJLP_DEC = 896;

__device__ __forceinline__ float sigf(float x) { return 1.f / (1.f + expf(-x)); }

union FragU { v16bf v; uint4 q[2]; };

// ---------------------------------------------------------------------------
// Fallback NT WMMA GEMM (direct global loads): C[M,N] = A[M,K] * B[N,K]^T
// One wave per 32x32 tile. M,N multiples of 32; K multiple of 32.
// ---------------------------------------------------------------------------
__global__ __launch_bounds__(32)
void k_gemm_nt(const __bf16* __restrict__ A, const __bf16* __restrict__ B,
               float* __restrict__ C, int K, int lda, int ldb, int ldc, int acc)
{
    const int tn   = blockIdx.x * 32;
    const int tm   = blockIdx.y * 32;
    const int lane = threadIdx.x;
    const int half = lane >> 4, r = lane & 15;

    v8f c00 = {0,0,0,0,0,0,0,0};
    v8f c01 = c00, c10 = c00, c11 = c00;
    if (acc) {
#pragma unroll
        for (int v = 0; v < 8; ++v) {
            const int row = v + 8 * half;
            c00[v] = C[(size_t)(tm + row) * ldc + tn + r];
            c01[v] = C[(size_t)(tm + row) * ldc + tn + 16 + r];
            c10[v] = C[(size_t)(tm + 16 + row) * ldc + tn + r];
            c11[v] = C[(size_t)(tm + 16 + row) * ldc + tn + 16 + r];
        }
    }
    const __bf16* a0p = A + (size_t)(tm + r) * lda;
    const __bf16* a1p = a0p + (size_t)16 * lda;
    const __bf16* b0p = B + (size_t)(tn + r) * ldb;
    const __bf16* b1p = b0p + (size_t)16 * ldb;

    for (int k0 = 0; k0 < K; k0 += 32) {
        const int o = k0 + half * 8;
        FragU a0, a1, b0, b1;
        a0.q[0] = *(const uint4*)(a0p + o); a0.q[1] = *(const uint4*)(a0p + o + 16);
        a1.q[0] = *(const uint4*)(a1p + o); a1.q[1] = *(const uint4*)(a1p + o + 16);
        b0.q[0] = *(const uint4*)(b0p + o); b0.q[1] = *(const uint4*)(b0p + o + 16);
        b1.q[0] = *(const uint4*)(b1p + o); b1.q[1] = *(const uint4*)(b1p + o + 16);
        c00 = __builtin_amdgcn_wmma_f32_16x16x32_bf16(false, a0.v, false, b0.v, (short)0, c00, false, false);
        c01 = __builtin_amdgcn_wmma_f32_16x16x32_bf16(false, a0.v, false, b1.v, (short)0, c01, false, false);
        c10 = __builtin_amdgcn_wmma_f32_16x16x32_bf16(false, a1.v, false, b0.v, (short)0, c10, false, false);
        c11 = __builtin_amdgcn_wmma_f32_16x16x32_bf16(false, a1.v, false, b1.v, (short)0, c11, false, false);
    }
#pragma unroll
    for (int v = 0; v < 8; ++v) {
        const int row = v + 8 * half;
        C[(size_t)(tm + row) * ldc + tn + r]           = c00[v];
        C[(size_t)(tm + row) * ldc + tn + 16 + r]      = c01[v];
        C[(size_t)(tm + 16 + row) * ldc + tn + r]      = c10[v];
        C[(size_t)(tm + 16 + row) * ldc + tn + 16 + r] = c11[v];
    }
}

// ---------------------------------------------------------------------------
// Main NT WMMA GEMM with async global->LDS double buffering.
// 4 waves / 128 threads per block; 64x64 block tile; K step 32.
// LDS tiles padded to 40 elements (80 B) per row -> conflict-free b128 reads.
// Each wave issues 4 GLOBAL_LOAD_ASYNC_TO_LDS_B128 per stage (2 KB), so
// "s_wait_asynccnt 4" after issuing stage k+1 guarantees stage k landed.
// Requires M%64==0, N%64==0, K%32==0.
// ---------------------------------------------------------------------------
#define LDSLD 40   // padded LDS row stride in bf16 elements

__device__ __forceinline__ void async_tile(const __bf16* __restrict__ G, int ld,
                                           int row0, unsigned ldsbase, int t)
{
#pragma unroll
    for (int i = 0; i < 2; ++i) {
        const int c = t + i * 128;         // 256 chunks of 16 B cover 64x32 bf16
        const int row = c >> 2;
        const int kc = (c & 3) * 8;
        unsigned la = ldsbase + (unsigned)(row * LDSLD + kc) * 2u;
        unsigned long long ga = (unsigned long long)(G + (size_t)(row0 + row) * ld + kc);
        asm volatile("global_load_async_to_lds_b128 %0, %1, off"
                     :: "v"(la), "v"(ga) : "memory");
    }
}

__global__ __launch_bounds__(128)
void k_gemm_nt_lds(const __bf16* __restrict__ A, const __bf16* __restrict__ B,
                   float* __restrict__ C, int K, int lda, int ldb, int ldc, int acc)
{
    __shared__ __align__(16) __bf16 sA[2][64 * LDSLD];
    __shared__ __align__(16) __bf16 sB[2][64 * LDSLD];

    const int tn   = blockIdx.x * 64;
    const int tm   = blockIdx.y * 64;
    const int t    = threadIdx.x;
    const int lane = t & 31;
    const int wave = t >> 5;
    const int half = lane >> 4, r = lane & 15;
    const int mr   = (wave >> 1) << 5;   // 0 / 32 quadrant row
    const int ncq  = (wave & 1) << 5;    // 0 / 32 quadrant col

    v8f c00 = {0,0,0,0,0,0,0,0};
    v8f c01 = c00, c10 = c00, c11 = c00;
    if (acc) {
#pragma unroll
        for (int v = 0; v < 8; ++v) {
            const int row = tm + mr + v + 8 * half;
            c00[v] = C[(size_t)row * ldc + tn + ncq + r];
            c01[v] = C[(size_t)row * ldc + tn + ncq + 16 + r];
            c10[v] = C[(size_t)(row + 16) * ldc + tn + ncq + r];
            c11[v] = C[(size_t)(row + 16) * ldc + tn + ncq + 16 + r];
        }
    }

    const unsigned baseA = (unsigned)(unsigned long long)(&sA[0][0]);
    const unsigned baseB = (unsigned)(unsigned long long)(&sB[0][0]);
    const unsigned bufsz = (unsigned)(64 * LDSLD * 2);

    // prefetch stage 0
    async_tile(A, lda, tm, baseA, t);
    async_tile(B, ldb, tn, baseB, t);

    const int nk = K >> 5;
    for (int ks = 0; ks < nk; ++ks) {
        const int cur = ks & 1;
        if (ks + 1 < nk) {
            async_tile(A + (ks + 1) * 32, lda, tm, baseA + (cur ^ 1) * bufsz, t);
            async_tile(B + (ks + 1) * 32, ldb, tn, baseB + (cur ^ 1) * bufsz, t);
            asm volatile("s_wait_asynccnt 0x4" ::: "memory");
        } else {
            asm volatile("s_wait_asynccnt 0x0" ::: "memory");
        }
        __syncthreads();

        const __bf16* pa0 = &sA[cur][(mr + r) * LDSLD + half * 8];
        const __bf16* pa1 = pa0 + 16 * LDSLD;
        const __bf16* pb0 = &sB[cur][(ncq + r) * LDSLD + half * 8];
        const __bf16* pb1 = pb0 + 16 * LDSLD;
        FragU a0, a1, b0, b1;
        a0.q[0] = *(const uint4*)(pa0); a0.q[1] = *(const uint4*)(pa0 + 16);
        a1.q[0] = *(const uint4*)(pa1); a1.q[1] = *(const uint4*)(pa1 + 16);
        b0.q[0] = *(const uint4*)(pb0); b0.q[1] = *(const uint4*)(pb0 + 16);
        b1.q[0] = *(const uint4*)(pb1); b1.q[1] = *(const uint4*)(pb1 + 16);
        c00 = __builtin_amdgcn_wmma_f32_16x16x32_bf16(false, a0.v, false, b0.v, (short)0, c00, false, false);
        c01 = __builtin_amdgcn_wmma_f32_16x16x32_bf16(false, a0.v, false, b1.v, (short)0, c01, false, false);
        c10 = __builtin_amdgcn_wmma_f32_16x16x32_bf16(false, a1.v, false, b0.v, (short)0, c10, false, false);
        c11 = __builtin_amdgcn_wmma_f32_16x16x32_bf16(false, a1.v, false, b1.v, (short)0, c11, false, false);
        __syncthreads();
    }

#pragma unroll
    for (int v = 0; v < 8; ++v) {
        const int row = tm + mr + v + 8 * half;
        C[(size_t)row * ldc + tn + ncq + r]             = c00[v];
        C[(size_t)row * ldc + tn + ncq + 16 + r]        = c01[v];
        C[(size_t)(row + 16) * ldc + tn + ncq + r]      = c10[v];
        C[(size_t)(row + 16) * ldc + tn + ncq + 16 + r] = c11[v];
    }
}

// ---------------------------------------------------------------------------
// Elementwise / small kernels
// ---------------------------------------------------------------------------
__global__ void k_uv_spatial(const float* __restrict__ X, const float* __restrict__ W,
                             __bf16* __restrict__ Ut)
{
    int tid = blockIdx.x * blockDim.x + threadIdx.x;
    if (tid >= B_ * T_ * N_ * HID_) return;
    int bt = tid / (N_ * HID_);
    int rem = tid % (N_ * HID_);
    int n = rem >> 6, h = rem & 63;
    const float* x = X + (size_t)(bt * N_ + n) * C_;
    float s = 0.f;
#pragma unroll
    for (int c = 0; c < C_; ++c) s += x[c] * W[c * HID_ + h];
    Ut[(size_t)n * (B_ * T_ * HID_) + bt * HID_ + h] = (__bf16)tanhf(3.f * s);
}

__global__ void k_uv_cat(const float* __restrict__ X, const float* __restrict__ W,
                         float* __restrict__ out)
{
    int tid = blockIdx.x * blockDim.x + threadIdx.x;
    if (tid >= B_ * T_ * C_ * HID_) return;
    int bt = tid >> 9;
    int rem = tid & 511;
    int c = rem >> 6, h = rem & 63;
    float s = 0.f;
    for (int n = 0; n < N_; ++n) s += X[(size_t)(bt * N_ + n) * C_ + c] * W[n * HID_ + h];
    out[tid] = tanhf(3.f * s);
}

__global__ void k_mm_small(const float* __restrict__ U, const float* __restrict__ V,
                           float* __restrict__ M)
{
    int tid = blockIdx.x * blockDim.x + threadIdx.x;
    if (tid >= 64) return;
    int c = tid >> 3, d = tid & 7;
    float s = 0.f;
    for (int r = 0; r < B_ * T_; ++r)
        for (int h = 0; h < HID_; ++h)
            s += U[(r * 8 + c) * 64 + h] * V[(r * 8 + d) * 64 + h];
    M[tid] = s;
}

__global__ void k_softmax_rows(const float* __restrict__ M, float* __restrict__ P, int n)
{
    int i = blockIdx.x * blockDim.x + threadIdx.x;
    if (i >= n) return;
    float mx = -1e30f;
    for (int j = 0; j < n; ++j)
        mx = fmaxf(mx, fmaxf(M[i * n + j] - M[j * n + i], 0.f));
    float s = 0.f;
    for (int j = 0; j < n; ++j)
        s += expf(fmaxf(M[i * n + j] - M[j * n + i], 0.f) - mx);
    float inv = 1.f / s;
    for (int j = 0; j < n; ++j)
        P[i * n + j] = expf(fmaxf(M[i * n + j] - M[j * n + i], 0.f) - mx) * inv;
}

__global__ void k_rowsum2(const float* __restrict__ A, const float* __restrict__ P,
                          float* __restrict__ nf, int n)
{
    int i = blockIdx.x * blockDim.x + threadIdx.x;
    if (i >= n) return;
    float sa = 0.f, sp = 0.f;
    for (int j = 0; j < n; ++j) { sa += A[i * n + j]; sp += P[i * n + j]; }
    nf[i * 2] = sa; nf[i * 2 + 1] = sp;
}

__global__ void k_xnf(const float* __restrict__ G, const float* __restrict__ nf,
                      float* __restrict__ out, int n)
{
    int i = blockIdx.x * blockDim.x + threadIdx.x;
    if (i >= n) return;
    float a0 = 0.f, a1 = 0.f;
    for (int j = 0; j < n; ++j) {
        float g = G[i * n + j];
        a0 += g * nf[j * 2]; a1 += g * nf[j * 2 + 1];
    }
    out[i * 2] = a0; out[i * 2 + 1] = a1;
}

__global__ void k_mlp1(const float* __restrict__ X2, const float* __restrict__ w1,
                       const float* __restrict__ b1, float* __restrict__ out, int n)
{
    int tid = blockIdx.x * blockDim.x + threadIdx.x;
    if (tid >= n * 256) return;
    int i = tid >> 8, j = tid & 255;
    float v = X2[i * 2] * w1[j] + X2[i * 2 + 1] * w1[256 + j] + b1[j];
    out[tid] = fmaxf(v, 0.f);
}

__global__ void k_mlp2_small(const float* __restrict__ h1, const float* __restrict__ w2,
                             const float* __restrict__ b2, float* __restrict__ out, int acc)
{
    int tid = blockIdx.x * blockDim.x + threadIdx.x;
    if (tid >= 8 * 256) return;
    int i = tid >> 8, j = tid & 255;
    float s = acc ? out[tid] : 0.f;
    s += b2[j];
    for (int k = 0; k < 256; ++k) s += h1[i * 256 + k] * w2[k * 256 + j];
    out[tid] = s;
}

__global__ void k_av_small(const float* __restrict__ key, const float* __restrict__ val,
                           float* __restrict__ alc)
{
    int tid = blockIdx.x * blockDim.x + threadIdx.x;
    if (tid >= 64) return;
    int i = tid >> 3, j = tid & 7;
    float s = 0.f;
    for (int q = 0; q < 256; ++q) s += key[i * 256 + q] * val[j * 256 + q];
    alc[tid] = s;
}

__global__ void k_fuse_graph(const float* __restrict__ alog, const float* __restrict__ A,
                             const float* __restrict__ P, float* __restrict__ G, int n)
{
    int tid = blockIdx.x * blockDim.x + threadIdx.x;
    if (tid >= n) return;
    float g = sigf(alog[tid]);
    G[tid] = g * A[tid] + (1.f - g) * P[tid];
}

__global__ void k_cast(const float* __restrict__ in, __bf16* __restrict__ out, int n)
{
    int tid = blockIdx.x * blockDim.x + threadIdx.x;
    if (tid >= n) return;
    out[tid] = (__bf16)in[tid];
}

__global__ void k_tcast(const float* __restrict__ in, __bf16* __restrict__ out,
                        int R, int Ccol, int ldo)
{
    int tid = blockIdx.x * blockDim.x + threadIdx.x;
    if (tid >= R * Ccol) return;
    int r = tid / Ccol, c = tid % Ccol;
    out[(size_t)c * ldo + r] = (__bf16)in[tid];
}

__global__ void k_cast_bias2(const float* __restrict__ in, const float* __restrict__ ba,
                             const float* __restrict__ bb, __bf16* __restrict__ out,
                             int cols, int n)
{
    int tid = blockIdx.x * blockDim.x + threadIdx.x;
    if (tid >= n) return;
    int j = tid % cols;
    out[tid] = (__bf16)(in[tid] + ba[j] + bb[j]);
}

__global__ void k_gst_set(const float* __restrict__ Gs, const float* __restrict__ Gs2,
                          __bf16* __restrict__ out)
{
    int tid = blockIdx.x * blockDim.x + threadIdx.x;
    if (tid >= 3 * N_ * N_) return;
    int k = tid / (N_ * N_);
    int rem = tid % (N_ * N_);
    int m = rem >> 9, n = rem & 511;
    float v;
    if (k == 0)      v = (m == n) ? 1.f : 0.f;
    else if (k == 1) v = Gs[n * N_ + m];
    else             v = 2.f * Gs2[n * N_ + m] - ((m == n) ? 1.f : 0.f);
    out[tid] = (__bf16)v;
}

__global__ void k_gcset(const float* __restrict__ Gc, float* __restrict__ out)
{
    int tid = blockIdx.x * blockDim.x + threadIdx.x;
    if (tid >= 128) return;
    int j = tid >> 6, rem = tid & 63;
    int c = rem >> 3, d = rem & 7;
    out[tid] = j ? Gc[rem] : ((c == d) ? 1.f : 0.f);
}

__global__ void k_build_xht(const float* __restrict__ Xp, int sxb, int sxn, int sxc, int WX,
                            const float* __restrict__ Hm, const float* __restrict__ ft,
                            __bf16* __restrict__ XHT, int L, int CLp)
{
    int tid = blockIdx.x * blockDim.x + threadIdx.x;
    int tot = B_ * N_ * C_ * L;
    if (tid >= tot) return;
    int b = tid / (N_ * C_ * L);
    int rem = tid % (N_ * C_ * L);
    int n = rem / (C_ * L);
    int rem2 = rem % (C_ * L);
    int c = rem2 / L, l = rem2 % L;
    float v;
    if (l < WX)                 v = Xp[(size_t)b * sxb + (size_t)n * sxn + c * sxc + l];
    else if (l < WX + HID_)     v = Hm[(((size_t)(b * N_ + n)) * C_ + c) * HID_ + (l - WX)];
    else                        v = ft[n * FEAT_ + (l - WX - HID_)];
    XHT[((size_t)b * CLp + (c * L + l)) * N_ + n] = (__bf16)v;
}

__global__ void k_t2feat(const float* __restrict__ t1, const float* __restrict__ Gcset,
                         __bf16* __restrict__ f, int L, int CLp, int KJL, int KJLp)
{
    int tid = blockIdx.x * blockDim.x + threadIdx.x;
    if (tid >= ROWS_ * KJL) return;
    int row = tid / KJL, q = tid % KJL;
    int k = q / (2 * L);
    int r2 = q % (2 * L);
    int j = r2 / L, l = r2 % L;
    int b = row >> 12, m = (row >> 3) & 511, d = row & 7;
    const float* tp = t1 + ((size_t)((k * 4 + b) * N_ + m)) * CLp + l;
    const float* gp = Gcset + j * 64 + d;
    float s = 0.f;
#pragma unroll
    for (int c = 0; c < C_; ++c) s += tp[c * L] * gp[c * 8];
    f[(size_t)row * KJLp + q] = (__bf16)s;
}

__global__ void k_zero_pad(__bf16* __restrict__ f, int KJL, int KJLp, int n)
{
    int tid = blockIdx.x * blockDim.x + threadIdx.x;
    if (tid >= n) return;
    int pad = KJLp - KJL;
    int row = tid / pad, jj = tid % pad;
    f[(size_t)row * KJLp + KJL + jj] = (__bf16)0.f;
}

__global__ void k_gate(const float* __restrict__ uv, const float* __restrict__ gb,
                       const float* __restrict__ H, float* __restrict__ upd,
                       float* __restrict__ rH)
{
    int tid = blockIdx.x * blockDim.x + threadIdx.x;
    if (tid >= ROWS_ * HID_) return;
    int row = tid >> 6, h = tid & 63;
    float u = sigf(uv[(size_t)row * 128 + h] + gb[h]);
    float r = sigf(uv[(size_t)row * 128 + 64 + h] + gb[64 + h]);
    upd[tid] = u;
    rH[tid] = r * H[tid];
}

__global__ void k_hupdate(const float* __restrict__ cpre, const float* __restrict__ cb,
                          const float* __restrict__ upd, float* __restrict__ H)
{
    int tid = blockIdx.x * blockDim.x + threadIdx.x;
    if (tid >= ROWS_ * HID_) return;
    int h = tid & 63;
    float cd = tanhf(cpre[tid] + cb[h]);
    float u = upd[tid];
    H[tid] = (1.f - u) * H[tid] + u * cd;
}

__global__ void k_out2(const float* __restrict__ o1, const float* __restrict__ b1,
                       const float* __restrict__ w2, const float* __restrict__ b2,
                       float* __restrict__ out, int step)
{
    int tid = blockIdx.x * blockDim.x + threadIdx.x;
    if (tid >= ROWS_) return;
    int b = tid >> 12, n = (tid >> 3) & 511, c = tid & 7;
    float s = b2[0];
#pragma unroll
    for (int j = 0; j < 32; ++j)
        s += fmaxf(o1[(size_t)tid * 32 + j] + b1[j], 0.f) * w2[j];
    out[((size_t)(b * 3 + step) * N_ + n) * C_ + c] = s;
}

// ---------------------------------------------------------------------------
// Host orchestration
// ---------------------------------------------------------------------------
static inline void gemm_nt(const void* A, const void* B, void* C, int M, int N, int K,
                           int lda, int ldb, int ldc, int acc, hipStream_t s)
{
    if ((M % 64) == 0 && (N % 64) == 0) {
        dim3 g((unsigned)(N / 64), (unsigned)(M / 64));
        k_gemm_nt_lds<<<g, 128, 0, s>>>((const __bf16*)A, (const __bf16*)B, (float*)C,
                                        K, lda, ldb, ldc, acc);
    } else {
        dim3 g((unsigned)(N / 32), (unsigned)(M / 32));
        k_gemm_nt<<<g, 32, 0, s>>>((const __bf16*)A, (const __bf16*)B, (float*)C,
                                   K, lda, ldb, ldc, acc);
    }
}

extern "C" void kernel_launch(void* const* d_in, const int* in_sizes, int n_in,
                              void* d_out, int out_size, void* d_ws, size_t ws_size,
                              hipStream_t stream)
{
    (void)in_sizes; (void)n_in; (void)out_size; (void)ws_size;
    const float* X     = (const float*)d_in[0];
    const float* As    = (const float*)d_in[1];
    const float* Ac    = (const float*)d_in[2];
    const float* ftIn  = (const float*)d_in[3];
    const float* WuS   = (const float*)d_in[4];
    const float* WvS   = (const float*)d_in[5];
    const float* WuC   = (const float*)d_in[6];
    const float* WvC   = (const float*)d_in[7];
    const float* mfSw1 = (const float*)d_in[8];
    const float* mfSb1 = (const float*)d_in[9];
    const float* mfSw2 = (const float*)d_in[10];
    const float* mfSb2 = (const float*)d_in[11];
    const float* mfCw1 = (const float*)d_in[12];
    const float* mfCb1 = (const float*)d_in[13];
    const float* mfCw2 = (const float*)d_in[14];
    const float* mfCb2 = (const float*)d_in[15];
    const float* egW = (const float*)d_in[16];
    const float* egb = (const float*)d_in[17];
    const float* ecW = (const float*)d_in[18];
    const float* ecb = (const float*)d_in[19];
    const float* dgW = (const float*)d_in[20];
    const float* dgb = (const float*)d_in[21];
    const float* dcW = (const float*)d_in[22];
    const float* dcb = (const float*)d_in[23];
    const float* ow1 = (const float*)d_in[24];
    const float* ob1 = (const float*)d_in[25];
    const float* ow2 = (const float*)d_in[26];
    const float* ob2 = (const float*)d_in[27];
    float* out = (float*)d_out;

    size_t off = 0;
    auto alloc = [&](size_t bytes) {
        char* p = (char*)d_ws + off;
        off = (off + bytes + 255) & ~(size_t)255;
        return (void*)p;
    };
    __bf16* Ub     = (__bf16*)alloc((size_t)N_ * 1536 * 2);
    __bf16* Vb     = (__bf16*)alloc((size_t)N_ * 1536 * 2);
    float*  Ms     = (float*)alloc((size_t)N_ * N_ * 4);
    float*  Ps     = (float*)alloc((size_t)N_ * N_ * 4);
    float*  nf     = (float*)alloc(N_ * 2 * 4);
    float*  An     = (float*)alloc(N_ * 2 * 4);
    float*  Pn     = (float*)alloc(N_ * 2 * 4);
    float*  h1f    = (float*)alloc((size_t)N_ * 256 * 4);
    __bf16* h1b    = (__bf16*)alloc((size_t)N_ * 256 * 2);
    __bf16* w2T    = (__bf16*)alloc((size_t)256 * 256 * 2);
    float*  keyf   = (float*)alloc((size_t)N_ * 256 * 4);
    float*  valf   = (float*)alloc((size_t)N_ * 256 * 4);
    __bf16* keyb   = (__bf16*)alloc((size_t)N_ * 256 * 2);
    __bf16* valb   = (__bf16*)alloc((size_t)N_ * 256 * 2);
    float*  alog   = (float*)alloc((size_t)N_ * N_ * 4);
    float*  Gs     = (float*)alloc((size_t)N_ * N_ * 4);
    __bf16* Gsb    = (__bf16*)alloc((size_t)N_ * N_ * 2);
    __bf16* GsTb   = (__bf16*)alloc((size_t)N_ * N_ * 2);
    float*  Gs2    = (float*)alloc((size_t)N_ * N_ * 4);
    __bf16* GsTset = (__bf16*)alloc((size_t)3 * N_ * N_ * 2);
    float*  Ucf    = (float*)alloc((size_t)192 * 64 * 4);
    float*  Vcf    = (float*)alloc((size_t)192 * 64 * 4);
    float*  Mc     = (float*)alloc(64 * 4);
    float*  Pc     = (float*)alloc(64 * 4);
    float*  nfc    = (float*)alloc(16 * 4);
    float*  Anc    = (float*)alloc(16 * 4);
    float*  Pnc    = (float*)alloc(16 * 4);
    float*  h1c    = (float*)alloc(8 * 256 * 4);
    float*  keyc   = (float*)alloc(8 * 256 * 4);
    float*  valc   = (float*)alloc(8 * 256 * 4);
    float*  alc    = (float*)alloc(64 * 4);
    float*  Gc     = (float*)alloc(64 * 4);
    float*  Gcset  = (float*)alloc(2 * 64 * 4);
    __bf16* WgTe   = (__bf16*)alloc((size_t)128 * KJLP_ENC * 2);
    __bf16* WcTe   = (__bf16*)alloc((size_t)64 * KJLP_ENC * 2);
    __bf16* WgTd   = (__bf16*)alloc((size_t)128 * KJLP_DEC * 2);
    __bf16* WcTd   = (__bf16*)alloc((size_t)64 * KJLP_DEC * 2);
    __bf16* w1T    = (__bf16*)alloc((size_t)32 * 64 * 2);
    __bf16* XHT    = (__bf16*)alloc((size_t)B_ * CLP_DEC * N_ * 2);
    float*  t1     = (float*)alloc((size_t)12 * N_ * CLP_DEC * 4);
    __bf16* featb  = (__bf16*)alloc((size_t)ROWS_ * KJLP_DEC * 2);
    float*  uv     = (float*)alloc((size_t)ROWS_ * 128 * 4);
    float*  upd    = (float*)alloc((size_t)ROWS_ * HID_ * 4);
    float*  rH     = (float*)alloc((size_t)ROWS_ * HID_ * 4);
    float*  cpre   = (float*)alloc((size_t)ROWS_ * HID_ * 4);
    float*  H      = (float*)alloc((size_t)ROWS_ * HID_ * 4);
    __bf16* Hbf    = (__bf16*)alloc((size_t)ROWS_ * HID_ * 2);
    float*  o1     = (float*)alloc((size_t)ROWS_ * 32 * 4);

    auto nb = [](int n) { return (unsigned)((n + 255) >> 8); };

    // ---- spatial adaptive graph ----
    k_uv_spatial<<<nb(B_*T_*N_*HID_), 256, 0, stream>>>(X, WuS, Ub);
    k_uv_spatial<<<nb(B_*T_*N_*HID_), 256, 0, stream>>>(X, WvS, Vb);
    gemm_nt(Ub, Vb, Ms, 512, 512, 1536, 1536, 1536, 512, 0, stream);
    k_softmax_rows<<<nb(512), 256, 0, stream>>>(Ms, Ps, 512);

    // ---- mixed fusion (spatial) ----
    k_rowsum2<<<nb(512), 256, 0, stream>>>(As, Ps, nf, 512);
    k_xnf<<<nb(512), 256, 0, stream>>>(As, nf, An, 512);
    k_xnf<<<nb(512), 256, 0, stream>>>(Ps, nf, Pn, 512);
    {
        const float* srcs[2] = {An, Pn};
        float* dsts[2] = {keyf, valf};
        for (int kv = 0; kv < 2; ++kv) {
            for (int p = 0; p < 2; ++p) {
                int widx = kv * 2 + p;
                k_mlp1<<<nb(512*256), 256, 0, stream>>>(srcs[p], mfSw1 + widx*512,
                                                        mfSb1 + widx*256, h1f, 512);
                k_cast<<<nb(512*256), 256, 0, stream>>>(h1f, h1b, 512*256);
                k_tcast<<<nb(256*256), 256, 0, stream>>>(mfSw2 + (size_t)widx*65536,
                                                         w2T, 256, 256, 256);
                gemm_nt(h1b, w2T, dsts[kv], 512, 256, 256, 256, 256, 256, p, stream);
            }
        }
    }
    k_cast_bias2<<<nb(512*256), 256, 0, stream>>>(keyf, mfSb2 + 0,   mfSb2 + 256, keyb, 256, 512*256);
    k_cast_bias2<<<nb(512*256), 256, 0, stream>>>(valf, mfSb2 + 512, mfSb2 + 768, valb, 256, 512*256);
    gemm_nt(keyb, valb, alog, 512, 512, 256, 256, 256, 512, 0, stream);
    k_fuse_graph<<<nb(512*512), 256, 0, stream>>>(alog, As, Ps, Gs, 512*512);

    // ---- Chebyshev (spatial, transposed set in bf16) ----
    k_cast<<<nb(512*512), 256, 0, stream>>>(Gs, Gsb, 512*512);
    k_tcast<<<nb(512*512), 256, 0, stream>>>(Gs, GsTb, 512, 512, 512);
    gemm_nt(Gsb, GsTb, Gs2, 512, 512, 512, 512, 512, 512, 0, stream);
    k_gst_set<<<nb(3*512*512), 256, 0, stream>>>(Gs, Gs2, GsTset);

    // ---- categorical branch (N=8, scalar kernels) ----
    k_uv_cat<<<nb(B_*T_*C_*HID_), 256, 0, stream>>>(X, WuC, Ucf);
    k_uv_cat<<<nb(B_*T_*C_*HID_), 256, 0, stream>>>(X, WvC, Vcf);
    k_mm_small<<<1, 64, 0, stream>>>(Ucf, Vcf, Mc);
    k_softmax_rows<<<1, 32, 0, stream>>>(Mc, Pc, 8);
    k_rowsum2<<<1, 32, 0, stream>>>(Ac, Pc, nfc, 8);
    k_xnf<<<1, 32, 0, stream>>>(Ac, nfc, Anc, 8);
    k_xnf<<<1, 32, 0, stream>>>(Pc, nfc, Pnc, 8);
    {
        const float* srcs[2] = {Anc, Pnc};
        float* dsts[2] = {keyc, valc};
        for (int kv = 0; kv < 2; ++kv) {
            for (int p = 0; p < 2; ++p) {
                int widx = kv * 2 + p;
                k_mlp1<<<nb(8*256), 256, 0, stream>>>(srcs[p], mfCw1 + widx*512,
                                                      mfCb1 + widx*256, h1c, 8);
                k_mlp2_small<<<nb(8*256), 256, 0, stream>>>(h1c, mfCw2 + (size_t)widx*65536,
                                                            mfCb2 + widx*256, dsts[kv], p);
            }
        }
    }
    k_av_small<<<1, 64, 0, stream>>>(keyc, valc, alc);
    k_fuse_graph<<<1, 64, 0, stream>>>(alc, Ac, Pc, Gc, 64);
    k_gcset<<<1, 128, 0, stream>>>(Gc, Gcset);

    // ---- weight prep (transpose + cast + zero-pad K) ----
    hipMemsetAsync(WgTe, 0, (size_t)128 * KJLP_ENC * 2, stream);
    k_tcast<<<nb(KJL_ENC*128), 256, 0, stream>>>(egW, WgTe, KJL_ENC, 128, KJLP_ENC);
    hipMemsetAsync(WcTe, 0, (size_t)64 * KJLP_ENC * 2, stream);
    k_tcast<<<nb(KJL_ENC*64), 256, 0, stream>>>(ecW, WcTe, KJL_ENC, 64, KJLP_ENC);
    hipMemsetAsync(WgTd, 0, (size_t)128 * KJLP_DEC * 2, stream);
    k_tcast<<<nb(KJL_DEC*128), 256, 0, stream>>>(dgW, WgTd, KJL_DEC, 128, KJLP_DEC);
    hipMemsetAsync(WcTd, 0, (size_t)64 * KJLP_DEC * 2, stream);
    k_tcast<<<nb(KJL_DEC*64), 256, 0, stream>>>(dcW, WcTd, KJL_DEC, 64, KJLP_DEC);
    k_tcast<<<nb(64*32), 256, 0, stream>>>(ow1, w1T, 64, 32, 64);

    hipMemsetAsync(H, 0, (size_t)ROWS_ * HID_ * 4, stream);

    auto run_bdg = [&](const float* Xp, int sxb, int sxn, int sxc, int WX,
                       const float* Hmid, const __bf16* WT, int Nout, float* Cout,
                       int L, int CLp, int KJL, int KJLp) {
        int CL = C_ * L;
        if (CL < CLp)
            hipMemsetAsync(XHT, 0, (size_t)B_ * CLp * N_ * 2, stream);
        k_build_xht<<<nb(B_*N_*C_*L), 256, 0, stream>>>(Xp, sxb, sxn, sxc, WX,
                                                        Hmid, ftIn, XHT, L, CLp);
        for (int k = 0; k < 3; ++k)
            for (int b = 0; b < 4; ++b)
                gemm_nt(GsTset + (size_t)k * N_ * N_,
                        XHT + (size_t)b * CLp * N_,
                        t1 + (size_t)((k * 4 + b) * N_) * CLp,
                        N_, CLp, N_, N_, N_, CLp, 0, stream);
        k_t2feat<<<nb(ROWS_*KJL), 256, 0, stream>>>(t1, Gcset, featb, L, CLp, KJL, KJLp);
        k_zero_pad<<<nb(ROWS_*(KJLp-KJL)), 256, 0, stream>>>(featb, KJL, KJLp, ROWS_*(KJLp-KJL));
        gemm_nt(featb, WT, Cout, ROWS_, Nout, KJLp, KJLp, KJLp, Nout, 0, stream);
    };

    auto run_cell = [&](const float* Xp, int sxb, int sxn, int sxc, int WX,
                        const __bf16* WgT, const float* gb,
                        const __bf16* WcT, const float* cb,
                        int L, int CLp, int KJL, int KJLp) {
        run_bdg(Xp, sxb, sxn, sxc, WX, H, WgT, 128, uv, L, CLp, KJL, KJLp);
        k_gate<<<nb(ROWS_*HID_), 256, 0, stream>>>(uv, gb, H, upd, rH);
        run_bdg(Xp, sxb, sxn, sxc, WX, rH, WcT, 64, cpre, L, CLp, KJL, KJLp);
        k_hupdate<<<nb(ROWS_*HID_), 256, 0, stream>>>(cpre, cb, upd, H);
    };

    for (int t = 0; t < T_; ++t)
        run_cell(X + (size_t)t * N_ * C_, T_ * N_ * C_, C_, 1, 1,
                 WgTe, egb, WcTe, ecb, L_ENC, CLP_ENC, KJL_ENC, KJLP_ENC);

    for (int st = 0; st < 3; ++st) {
        run_cell(H, N_ * C_ * HID_, C_ * HID_, HID_, HID_,
                 WgTd, dgb, WcTd, dcb, L_DEC, CLP_DEC, KJL_DEC, KJLP_DEC);
        k_cast<<<nb(ROWS_*HID_), 256, 0, stream>>>(H, Hbf, ROWS_*HID_);
        gemm_nt(Hbf, w1T, o1, ROWS_, 32, 64, 64, 64, 32, 0, stream);
        k_out2<<<nb(ROWS_), 256, 0, stream>>>(o1, ob1, ow2, ob2, out, st);
    }
}
